// TransXNet_SCRKD5_27264452395087
// MI455X (gfx1250) — matrix operand order repaired and checked by reading, not compiled, not run on hardware
//
#include <hip/hip_runtime.h>

// MI455X / gfx1250 — split-bf16 WMMA gram kernels for SCRKD loss.
// B=16, C=256, H=W=64.  Matrix-issue-bound (~26 GFLOP): route all gram work
// through v_wmma_f32_16x16x32_bf16 (16384 FLOP/instr, 8x the MACs of the f32
// 16x16x4 path) using the fp32-faithful hi/lo bf16 3-term decomposition:
//   x*y ~= hi(x)hi(y) + hi(x)lo(y) + lo(x)hi(y)   (fp32 accumulate)

typedef float v2f  __attribute__((ext_vector_type(2)));
typedef float v8f  __attribute__((ext_vector_type(8)));
typedef __bf16 v8bf  __attribute__((ext_vector_type(8)));
typedef __bf16 v16bf __attribute__((ext_vector_type(16)));

#define EPS_ 1e-8f
#define B_   16
#define C_   256
#define H_   64
#define W_   64
#define HW_  4096
#define CHW_ (C_ * HW_)
#define STRB 72   // bf16 LDS row stride: 144B rows -> every 16B chunk aligned

// workspace layout (floats)
#define OFF_SNCH 0
#define OFF_TNCH 4096
#define OFF_SNHE 8192
#define OFF_TNHE 9216
#define OFF_SNWI 10240
#define OFF_TNWI 11264
#define OFF_SGHW 12288                    // [16][2][64*64] student raw grams
#define OFF_TGHW (12288 + 131072)         // teacher raw grams
#define WS_FLOATS (12288 + 2 * 131072)

static __device__ inline v8f wmma_bf16(v16bf a, v16bf b, v8f c) {
    // (neg_a, A, neg_b, B, c_mod, C, reuse_a, reuse_b)
    return __builtin_amdgcn_wmma_f32_16x16x32_bf16(false, a, false, b, (short)0, c,
                                                   false, false);
}

// A-fragment: 16x32 bf16, lane m=l&15; lo-half lanes: v0-3=K0..7, v4-7=K16..23;
// hi-half lanes: v0-3=K8..15, v4-7=K24..31.  Two 16B loads per fragment.
static __device__ inline v16bf fragA(const __bf16* blk, int row, int kk, int hih) {
    const __bf16* p = blk + row * STRB + kk + (hih ? 8 : 0);
    v8bf p0 = *(const v8bf*)p;
    v8bf p1 = *(const v8bf*)(p + 16);
    return __builtin_shufflevector(p0, p1, 0, 1, 2, 3, 4, 5, 6, 7,
                                   8, 9, 10, 11, 12, 13, 14, 15);
}

// B-fragment: 32x16 bf16, lane n=l&15; lo-half lanes hold K=0..7,16..23, hi-half
// K=8..15,24..31 of row n (B[k][n] = Xj[n][k]).
static __device__ inline v16bf fragB(const __bf16* blk, int nrow, int kk, int hih) {
    const __bf16* p = blk + nrow * STRB + kk + (hih ? 8 : 0);
    v8bf p0 = *(const v8bf*)p;
    v8bf p1 = *(const v8bf*)(p + 16);
    return __builtin_shufflevector(p0, p1, 0, 1, 2, 3, 4, 5, 6, 7,
                                   8, 9, 10, 11, 12, 13, 14, 15);
}

// split one float4 into hi/lo bf16 and store 4 elements
static __device__ inline void cvt4(__bf16* hi, __bf16* lo, int pos, float4 v) {
    float f0 = v.x, f1 = v.y, f2 = v.z, f3 = v.w;
    __bf16 h0 = (__bf16)f0, h1 = (__bf16)f1, h2 = (__bf16)f2, h3 = (__bf16)f3;
    hi[pos + 0] = h0; hi[pos + 1] = h1; hi[pos + 2] = h2; hi[pos + 3] = h3;
    lo[pos + 0] = (__bf16)(f0 - (float)h0);
    lo[pos + 1] = (__bf16)(f1 - (float)h1);
    lo[pos + 2] = (__bf16)(f2 - (float)h2);
    lo[pos + 3] = (__bf16)(f3 - (float)h3);
}

// ---------------------------------------------------------------- zero init
__global__ void k_zero(float* __restrict__ ws, float* __restrict__ out, int out_n) {
    int i = blockIdx.x * 256 + threadIdx.x;
    if (i < WS_FLOATS) ws[i] = 0.0f;
    if (i < out_n) out[i] = 0.0f;
}

// ------------------------------------------------- norms (sum of squares)
__global__ __launch_bounds__(256) void k_norms(const float* __restrict__ s,
                                               const float* __restrict__ t,
                                               float* __restrict__ ws) {
    int bid = blockIdx.x;
    int sel = bid >> 12;             // 0 = student, 1 = teacher
    int bc  = bid & 4095;
    int b   = bc >> 8;
    const float* x = (sel == 0 ? s : t) + (size_t)bc * HW_;
    float* nch = ws + (sel == 0 ? OFF_SNCH : OFF_TNCH);
    float* nhe = ws + (sel == 0 ? OFF_SNHE : OFF_TNHE);
    float* nwi = ws + (sel == 0 ? OFF_SNWI : OFF_TNWI);

    __shared__ float sq[64 * 65];
    int tid = threadIdx.x;
    for (int i = 0; i < 16; ++i) {
        int e = tid + i * 256;
        float v = x[e];
        sq[(e >> 6) * 65 + (e & 63)] = v * v;
    }
    __syncthreads();
    float rs = 0.0f, cs = 0.0f;
    if (tid < 64) {
        for (int w = 0; w < 64; ++w) rs += sq[tid * 65 + w];
        for (int h = 0; h < 64; ++h) cs += sq[h * 65 + tid];
        atomicAdd(&nhe[b * 64 + tid], rs);
        atomicAdd(&nwi[b * 64 + tid], cs);
    }
    __syncthreads();
    if (tid < 64) sq[tid] = rs;
    __syncthreads();
    if (tid == 0) {
        float tot = 0.0f;
        for (int h = 0; h < 64; ++h) tot += sq[h];
        nch[bc] = tot;
    }
}

// ------------------------------------------------------- ch view (256x256)
// One block = one 64x64 macro-tile (I,J), I<=J (off-diag weighted x2), one
// batch, student+teacher fused.  4 waves, each owns a 16x64 strip.
__global__ __launch_bounds__(128) void k_gram_ch(const float* __restrict__ s,
                                                 const float* __restrict__ t,
                                                 const float* __restrict__ ws,
                                                 float* __restrict__ out) {
    int bid = blockIdx.x;            // 16 batches * 10 upper-tri macro pairs
    int b = bid / 10, p = bid % 10;
    int I = 0;
    {   int rem = p;
        while (rem >= 4 - I) { rem -= 4 - I; ++I; }
        p = I + rem;
    }
    int J = p;

    const float* sb = s + (size_t)b * CHW_;
    const float* tb = t + (size_t)b * CHW_;

    __shared__ __bf16 AsH[64 * STRB], AsL[64 * STRB], BsH[64 * STRB], BsL[64 * STRB];
    __shared__ __bf16 AtH[64 * STRB], AtL[64 * STRB], BtH[64 * STRB], BtL[64 * STRB];
    __shared__ float red[128];

    int tid = threadIdx.x, wave = tid >> 5, lane = tid & 31;
    int lrow = lane & 15;
    int hih  = lane >> 4;
    int hi8  = hih << 3;

    v8f accS[4], accT[4];
    const v8f vz = {};
    for (int j = 0; j < 4; ++j) { accS[j] = vz; accT[j] = vz; }

    for (int K0 = 0; K0 < HW_; K0 += 64) {
        for (int i2 = 0; i2 < 8; ++i2) {      // 1024 float4 / 128 threads
            int idx = tid + i2 * 128;
            int r = idx >> 4, c4 = (idx & 15) << 2;
            int pos = r * STRB + c4;
            float4 va = *(const float4*)(sb + (size_t)(I * 64 + r) * HW_ + K0 + c4);
            float4 vb = *(const float4*)(sb + (size_t)(J * 64 + r) * HW_ + K0 + c4);
            float4 vc = *(const float4*)(tb + (size_t)(I * 64 + r) * HW_ + K0 + c4);
            float4 vd = *(const float4*)(tb + (size_t)(J * 64 + r) * HW_ + K0 + c4);
            cvt4(AsH, AsL, pos, va);
            cvt4(BsH, BsL, pos, vb);
            cvt4(AtH, AtL, pos, vc);
            cvt4(BtH, BtL, pos, vd);
        }
        __syncthreads();
        int aRow = wave * 16 + lrow;
        #pragma unroll
        for (int kk = 0; kk < 64; kk += 32) {
            v16bf aSh = fragA(AsH, aRow, kk, hih);
            v16bf aSl = fragA(AsL, aRow, kk, hih);
            v16bf aTh = fragA(AtH, aRow, kk, hih);
            v16bf aTl = fragA(AtL, aRow, kk, hih);
            #pragma unroll
            for (int jt = 0; jt < 4; ++jt) {
                int nRow = jt * 16 + lrow;
                v16bf bSh = fragB(BsH, nRow, kk, hih);
                v16bf bSl = fragB(BsL, nRow, kk, hih);
                v16bf bTh = fragB(BtH, nRow, kk, hih);
                v16bf bTl = fragB(BtL, nRow, kk, hih);
                accS[jt] = wmma_bf16(aSh, bSh, accS[jt]);
                accT[jt] = wmma_bf16(aTh, bTh, accT[jt]);
                accS[jt] = wmma_bf16(aSh, bSl, accS[jt]);
                accT[jt] = wmma_bf16(aTh, bTl, accT[jt]);
                accS[jt] = wmma_bf16(aSl, bSh, accS[jt]);
                accT[jt] = wmma_bf16(aTl, bTh, accT[jt]);
            }
        }
        __syncthreads();
    }

    // fused normalize + squared-diff
    const float* sn = ws + OFF_SNCH + b * 256;
    const float* tn = ws + OFF_TNCH + b * 256;
    float wgt = (I == J) ? 1.0f : 2.0f;
    float lsum = 0.0f;
    for (int jt = 0; jt < 4; ++jt) {
        int j = J * 64 + jt * 16 + lrow;
        float snj = sqrtf(sn[j]), tnj = sqrtf(tn[j]);
        #pragma unroll
        for (int v = 0; v < 8; ++v) {
            int i = I * 64 + wave * 16 + v + hi8;
            float sni = sqrtf(sn[i]), tni = sqrtf(tn[i]);
            float sv = accS[jt][v] / fmaxf(sni * snj, EPS_);
            float tv = accT[jt][v] / fmaxf(tni * tnj, EPS_);
            float d = sv - tv;
            lsum += wgt * d * d;
        }
    }
    red[tid] = lsum;
    __syncthreads();
    for (int st = 64; st > 0; st >>= 1) {
        if (tid < st) red[tid] += red[tid + st];
        __syncthreads();
    }
    if (tid == 0) atomicAdd(out, red[0] * (1.0f / (16.0f * 256.0f * 256.0f)));
}

// ------------------------------------------- he/wi views (64x64, K=16384)
// K-chunk of 64 == one contiguous [b,c] 64x64 slab; he stores straight, wi
// transposed.  K split over 8 segments; raw grams accumulated atomically.
__global__ __launch_bounds__(128) void k_gram_hw(const float* __restrict__ s,
                                                 const float* __restrict__ t,
                                                 float* __restrict__ ws) {
    int bid  = blockIdx.x;           // b*16 + view*8 + seg
    int b    = bid >> 4;
    int view = (bid >> 3) & 1;       // 0 = he, 1 = wi
    int seg  = bid & 7;
    const float* sb = s + (size_t)b * CHW_;
    const float* tb = t + (size_t)b * CHW_;
    float* sG = ws + OFF_SGHW + (size_t)(b * 2 + view) * 4096;
    float* tG = ws + OFF_TGHW + (size_t)(b * 2 + view) * 4096;

    __shared__ __bf16 SsH[64 * STRB], SsL[64 * STRB];
    __shared__ __bf16 StH[64 * STRB], StL[64 * STRB];
    int tid = threadIdx.x, wave = tid >> 5, lane = tid & 31;
    int lrow = lane & 15;
    int hih  = lane >> 4;
    int hi8  = hih << 3;

    v8f accS[4], accT[4];
    const v8f vz = {};
    for (int j = 0; j < 4; ++j) { accS[j] = vz; accT[j] = vz; }

    for (int c = seg * 32; c < seg * 32 + 32; ++c) {
        const float* gs = sb + (size_t)c * HW_;
        const float* gt = tb + (size_t)c * HW_;
        __syncthreads();
        for (int i2 = 0; i2 < 32; ++i2) {
            int e = tid + i2 * 128;
            int h = e >> 6, w = e & 63;
            int pos = (view == 0) ? h * STRB + w : w * STRB + h;
            float fs = gs[e], ft = gt[e];
            __bf16 hs = (__bf16)fs, ht = (__bf16)ft;
            SsH[pos] = hs; SsL[pos] = (__bf16)(fs - (float)hs);
            StH[pos] = ht; StL[pos] = (__bf16)(ft - (float)ht);
        }
        __syncthreads();
        int aRow = wave * 16 + lrow;
        #pragma unroll
        for (int kk = 0; kk < 64; kk += 32) {
            v16bf aSh = fragA(SsH, aRow, kk, hih);
            v16bf aSl = fragA(SsL, aRow, kk, hih);
            v16bf aTh = fragA(StH, aRow, kk, hih);
            v16bf aTl = fragA(StL, aRow, kk, hih);
            #pragma unroll
            for (int jt = 0; jt < 4; ++jt) {
                int nRow = jt * 16 + lrow;
                v16bf bSh = fragB(SsH, nRow, kk, hih);
                v16bf bSl = fragB(SsL, nRow, kk, hih);
                v16bf bTh = fragB(StH, nRow, kk, hih);
                v16bf bTl = fragB(StL, nRow, kk, hih);
                accS[jt] = wmma_bf16(aSh, bSh, accS[jt]);
                accT[jt] = wmma_bf16(aTh, bTh, accT[jt]);
                accS[jt] = wmma_bf16(aSh, bSl, accS[jt]);
                accT[jt] = wmma_bf16(aTh, bTl, accT[jt]);
                accS[jt] = wmma_bf16(aSl, bSh, accS[jt]);
                accT[jt] = wmma_bf16(aTl, bTh, accT[jt]);
            }
        }
    }
    for (int jt = 0; jt < 4; ++jt) {
        int j = jt * 16 + lrow;
        #pragma unroll
        for (int v = 0; v < 8; ++v) {
            int i = wave * 16 + v + hi8;
            atomicAdd(&sG[i * 64 + j], accS[jt][v]);
            atomicAdd(&tG[i * 64 + j], accT[jt][v]);
        }
    }
}

// -------------------------------------- normalize he/wi grams + MSE reduce
__global__ __launch_bounds__(256) void k_mse_hw(const float* __restrict__ ws,
                                                float* __restrict__ out) {
    int bid = blockIdx.x;            // b*2 + view
    int b = bid >> 1, view = bid & 1;
    const float* sG = ws + OFF_SGHW + (size_t)bid * 4096;
    const float* tG = ws + OFF_TGHW + (size_t)bid * 4096;
    const float* sn = ws + (view == 0 ? OFF_SNHE : OFF_SNWI) + b * 64;
    const float* tn = ws + (view == 0 ? OFF_TNHE : OFF_TNWI) + b * 64;
    __shared__ float red[256];
    int tid = threadIdx.x;
    float lsum = 0.0f;
    for (int i2 = 0; i2 < 16; ++i2) {
        int e = tid + i2 * 256;
        int i = e >> 6, j = e & 63;
        float sv = sG[e] / fmaxf(sqrtf(sn[i]) * sqrtf(sn[j]), EPS_);
        float tv = tG[e] / fmaxf(sqrtf(tn[i]) * sqrtf(tn[j]), EPS_);
        float d = sv - tv;
        lsum += d * d;
    }
    red[tid] = lsum;
    __syncthreads();
    for (int st = 128; st > 0; st >>= 1) {
        if (tid < st) red[tid] += red[tid + st];
        __syncthreads();
    }
    if (tid == 0) atomicAdd(out, red[0] * (1.0f / (16.0f * 64.0f * 64.0f)));
}

// -------------------------------------------------------------- KD (preds)
__global__ __launch_bounds__(256) void k_kd(const float* __restrict__ spred,
                                            const float* __restrict__ tpred,
                                            float* __restrict__ out) {
    __shared__ float red[256];
    int tid = threadIdx.x;
    int e = blockIdx.x * 256 + tid;
    float x = spred[e], y = tpred[e];
    float a  = 1.0f / (1.0f + expf(-x));
    float bb = 1.0f / (1.0f + expf(-y));
    float p1 = 1.0f / (1.0f + expf(-(2.0f * a - 1.0f)));
    float s1 = 1.0f / (1.0f + expf(-(2.0f * bb - 1.0f)));
    float p2 = 1.0f - p1, s2 = 1.0f - s1;
    float term = 0.5f * (s1 * (logf(s1) - logf(p1)) +
                         s2 * (logf(s2) - logf(p2)));
    red[tid] = term;
    __syncthreads();
    for (int st = 128; st > 0; st >>= 1) {
        if (tid < st) red[tid] += red[tid + st];
        __syncthreads();
    }
    if (tid == 0) atomicAdd(out, red[0] * (4.0f / 131072.0f));
}

extern "C" void kernel_launch(void* const* d_in, const int* in_sizes, int n_in,
                              void* d_out, int out_size, void* d_ws, size_t ws_size,
                              hipStream_t stream) {
    const float* s_feat = (const float*)d_in[0];
    const float* t_feat = (const float*)d_in[1];
    const float* s_pred = (const float*)d_in[2];
    const float* t_pred = (const float*)d_in[3];
    float* out = (float*)d_out;
    float* ws  = (float*)d_ws;

    k_zero<<<(WS_FLOATS + 255) / 256, 256, 0, stream>>>(ws, out, out_size);
    k_norms<<<2 * B_ * C_, 256, 0, stream>>>(s_feat, t_feat, ws);
    k_gram_ch<<<B_ * 10, 128, 0, stream>>>(s_feat, t_feat, ws, out);
    k_gram_hw<<<B_ * 2 * 8, 128, 0, stream>>>(s_feat, t_feat, ws);
    k_mse_hw<<<B_ * 2, 256, 0, stream>>>(ws, out);
    k_kd<<<(B_ * H_ * W_) / 256, 256, 0, stream>>>(s_pred, t_pred, out);
}
